// TSPGNN_81853486727223
// MI455X (gfx1250) — matrix operand order repaired and checked by reading, not compiled, not run on hardware
//
#include <hip/hip_runtime.h>
#include <hip/hip_bf16.h>

#define H 128
#define WAVES 8

typedef __attribute__((ext_vector_type(16))) __bf16 v16bf;
typedef __attribute__((ext_vector_type(8)))  __bf16 v8bf;
typedef __attribute__((ext_vector_type(8)))  float  v8f;

static __device__ __forceinline__ v16bf cat16(v8bf lo, v8bf hi) {
    return __builtin_shufflevector(lo, hi, 0,1,2,3,4,5,6,7,8,9,10,11,12,13,14,15);
}

// A fragment (16x32 bf16) from a row-major [16][128] tile; rowbase already points at row m.
// Lane<16 owns K {0..7, 16..23} of the chunk, lane>=16 owns {8..15, 24..31} (ISA 7.12.2).
static __device__ __forceinline__ v16bf load_afrag(const __bf16* rowbase, int kc, int kbA) {
    const __bf16* p = rowbase + kc * 32 + kbA;
    return cat16(*(const v8bf*)p, *(const v8bf*)(p + 16));
}

// B fragment (32x16 bf16) from transposed weights Wt[N][K=128]; lane owns col n = lane&15,
// lanes 0-15 hold K {0..7, 16..23} of the chunk, lanes 16-31 hold K {8..15, 24..31} (matches A).
static __device__ __forceinline__ v16bf load_bfragT(const __bf16* Wt, int nt, int kc, int lane) {
    const __bf16* p = Wt + (size_t)(nt * 16 + (lane & 15)) * H + kc * 32 + ((lane < 16) ? 0 : 8);
    return cat16(*(const v8bf*)p, *(const v8bf*)(p + 16));
}

static __device__ __forceinline__ v8f wmma_bf16(v16bf a, v16bf b, v8f c) {
    return __builtin_amdgcn_wmma_f32_16x16x32_bf16(false, a, false, b, (short)0, c, false, false);
}

static __device__ __forceinline__ void wave_lds_fence() {
    asm volatile("s_wait_dscnt 0x0" ::: "memory");
}

// ---------------- small utility kernels ----------------

__global__ void k_cvt_transpose(const float* __restrict__ W, __bf16* __restrict__ Wt,
                                int K, int Nc) {
    int idx = blockIdx.x * blockDim.x + threadIdx.x;
    if (idx >= K * Nc) return;
    int k = idx / Nc, n = idx % Nc;
    Wt[(size_t)n * K + k] = (__bf16)W[idx];
}

__global__ void k_deg_init(float* __restrict__ deg, int N) {
    int i = blockIdx.x * blockDim.x + threadIdx.x;
    if (i < N) deg[i] = 1.0f;   // self-loop
}

__global__ void k_deg_edges(const int* __restrict__ col, float* __restrict__ deg, int E) {
    int e = blockIdx.x * blockDim.x + threadIdx.x;
    if (e < E) atomicAdd(&deg[col[e]], 1.0f);
}

__global__ void k_dinv(const float* __restrict__ deg, float* __restrict__ dinv, int N) {
    int i = blockIdx.x * blockDim.x + threadIdx.x;
    if (i < N) { float d = deg[i]; dinv[i] = d > 0.f ? rsqrtf(d) : 0.f; }
}

// layer 1 transform: x[N,8] @ W1[8,128]
__global__ void k_gemm_in8(const float* __restrict__ x, const float* __restrict__ W1,
                           float* __restrict__ outm, int N) {
    int idx = blockIdx.x * blockDim.x + threadIdx.x;
    if (idx >= N * H) return;
    int i = idx >> 7, j = idx & (H - 1);
    const float* xr = x + (size_t)i * 8;
    float s = 0.f;
    #pragma unroll
    for (int f = 0; f < 8; ++f) s += xr[f] * W1[f * H + j];
    outm[idx] = s;
}

// node GEMM: tmp[N,128] = h_bf[N,128] @ W (via transposed bf16 Wt[128][128]); WMMA
__global__ __launch_bounds__(256) void k_node_gemm(const __bf16* __restrict__ hb,
                                                   const __bf16* __restrict__ Wt,
                                                   float* __restrict__ outm, int N) {
    int wave = threadIdx.x >> 5, lane = threadIdx.x & 31;
    int row0 = blockIdx.x * 16;
    if (row0 >= N) return;
    int m = lane & 15, kbA = (lane < 16) ? 0 : 8;
    const __bf16* arow = hb + (size_t)(row0 + m) * H;
    v16bf A[4];
    #pragma unroll
    for (int kc = 0; kc < 4; ++kc) A[kc] = load_afrag(arow, kc, kbA);
    int nt = wave;  // 8 waves cover 8 column tiles
    v8f c = {0.f, 0.f, 0.f, 0.f, 0.f, 0.f, 0.f, 0.f};
    #pragma unroll
    for (int kc = 0; kc < 4; ++kc) c = wmma_bf16(A[kc], load_bfragT(Wt, nt, kc, lane), c);
    int roff = (lane < 16) ? 0 : 8;
    #pragma unroll
    for (int r = 0; r < 8; ++r)
        outm[(size_t)(row0 + r + roff) * H + nt * 16 + m] = c[r];
}

// out = dinv[i]^2 * tmp + bias (self-loop contribution + bias init)
__global__ void k_selfinit(const float* __restrict__ tmp, const float* __restrict__ dinv,
                           const float* __restrict__ b, float* __restrict__ outm, int N) {
    int idx = blockIdx.x * blockDim.x + threadIdx.x;
    if (idx >= N * H) return;
    int i = idx >> 7, j = idx & (H - 1);
    float di = dinv[i];
    outm[idx] = di * di * tmp[idx] + b[j];
}

// scatter: out[col] += dinv[row]*dinv[col] * tmp[row]  (4 floats per thread)
__global__ void k_scatter(const float* __restrict__ tmp, const float* __restrict__ dinv,
                          const int* __restrict__ row, const int* __restrict__ col,
                          float* __restrict__ outm, int E) {
    int tid = blockIdx.x * blockDim.x + threadIdx.x;
    if (tid >= E * 32) return;
    int e = tid >> 5, j = (tid & 31) * 4;
    int r = row[e], c = col[e];
    float w = dinv[r] * dinv[c];
    float4 v = *(const float4*)(tmp + (size_t)r * H + j);
    float* dst = outm + (size_t)c * H + j;
    atomicAdd(dst + 0, w * v.x);
    atomicAdd(dst + 1, w * v.y);
    atomicAdd(dst + 2, w * v.z);
    atomicAdd(dst + 3, w * v.w);
}

__global__ void k_relu_cvt(float* __restrict__ h, __bf16* __restrict__ hb, int n) {
    int idx = blockIdx.x * blockDim.x + threadIdx.x;
    if (idx >= n) return;
    float v = fmaxf(h[idx], 0.f);
    h[idx] = v;
    hb[idx] = (__bf16)v;
}

// ---------------- fused edge encoder + classifier head ----------------
__global__ __launch_bounds__(256) void k_edge_head(
    const int* __restrict__ row, const int* __restrict__ col,
    const float* __restrict__ ea, const __bf16* __restrict__ hb,
    const float* __restrict__ We1, const float* __restrict__ be1,
    const __bf16* __restrict__ We2t, const float* __restrict__ be2,
    const __bf16* __restrict__ Wc1at, const __bf16* __restrict__ Wc1bt,
    const __bf16* __restrict__ Wc1ct, const float* __restrict__ bc1,
    const __bf16* __restrict__ Wc2t, const float* __restrict__ bc2,
    const float* __restrict__ Wc3, const float* __restrict__ bc3,
    float* __restrict__ outm, int E) {
    __shared__ __align__(16) __bf16 sbuf[WAVES * 16 * H];  // 4 KB per wave, reused 3x
    int wave = threadIdx.x >> 5, lane = threadIdx.x & 31;
    int tile = blockIdx.x * WAVES + wave;
    int e0 = tile * 16;
    if (e0 >= E) return;
    int m = lane & 15;
    int e = e0 + m;
    int kbA = (lane < 16) ? 0 : 8;
    int roff = (lane < 16) ? 0 : 8;
    __bf16* my = sbuf + wave * 16 * H;

    // ---- edge MLP layer 1 (K=4, VALU) directly into A fragments ----
    float4 a4 = *(const float4*)(ea + (size_t)e * 4);
    v16bf tA[4];
    #pragma unroll
    for (int kc = 0; kc < 4; ++kc) {
        v16bf t;
        #pragma unroll
        for (int i = 0; i < 16; ++i) {
            int k = kc * 32 + kbA + ((i >> 3) << 4) + (i & 7);
            float v = a4.x * We1[k] + a4.y * We1[H + k] + a4.z * We1[2 * H + k] +
                      a4.w * We1[3 * H + k] + be1[k];
            t[i] = (__bf16)fmaxf(v, 0.f);
        }
        tA[kc] = t;
    }
    // ---- ef = relu1 @ We2 + be2 -> LDS tile (bf16, D->row-major) ----
    #pragma unroll
    for (int nt = 0; nt < 8; ++nt) {
        float bi = be2[nt * 16 + m];
        v8f c = {bi, bi, bi, bi, bi, bi, bi, bi};
        #pragma unroll
        for (int kc = 0; kc < 4; ++kc) c = wmma_bf16(tA[kc], load_bfragT(We2t, nt, kc, lane), c);
        #pragma unroll
        for (int r = 0; r < 8; ++r) my[(r + roff) * H + nt * 16 + m] = (__bf16)c[r];
    }
    wave_lds_fence();
    v16bf efA[4];
    #pragma unroll
    for (int kc = 0; kc < 4; ++kc) efA[kc] = load_afrag(my + m * H, kc, kbA);

    // ---- gather endpoint node features as A fragments ----
    int rn = row[e], cn = col[e];
    const __bf16* hr = hb + (size_t)rn * H;
    const __bf16* hc = hb + (size_t)cn * H;
    v16bf hrA[4], hcA[4];
    #pragma unroll
    for (int kc = 0; kc < 4; ++kc) { hrA[kc] = load_afrag(hr, kc, kbA); hcA[kc] = load_afrag(hc, kc, kbA); }
    wave_lds_fence();

    // ---- z1 = hr@Wc1a + hc@Wc1b + ef@Wc1c + bc1, relu -> LDS (bf16) ----
    #pragma unroll
    for (int nt = 0; nt < 8; ++nt) {
        float bi = bc1[nt * 16 + m];
        v8f c = {bi, bi, bi, bi, bi, bi, bi, bi};
        #pragma unroll
        for (int kc = 0; kc < 4; ++kc) c = wmma_bf16(hrA[kc], load_bfragT(Wc1at, nt, kc, lane), c);
        #pragma unroll
        for (int kc = 0; kc < 4; ++kc) c = wmma_bf16(hcA[kc], load_bfragT(Wc1bt, nt, kc, lane), c);
        #pragma unroll
        for (int kc = 0; kc < 4; ++kc) c = wmma_bf16(efA[kc], load_bfragT(Wc1ct, nt, kc, lane), c);
        #pragma unroll
        for (int r = 0; r < 8; ++r) my[(r + roff) * H + nt * 16 + m] = (__bf16)fmaxf(c[r], 0.f);
    }
    wave_lds_fence();
    v16bf z1A[4];
    #pragma unroll
    for (int kc = 0; kc < 4; ++kc) z1A[kc] = load_afrag(my + m * H, kc, kbA);
    wave_lds_fence();

    // ---- z2 = z1 @ Wc2 + bc2, relu -> LDS (f32, 16x64 = same 4KB) ----
    float* my2 = (float*)my;
    #pragma unroll
    for (int nt = 0; nt < 4; ++nt) {
        float bi = bc2[nt * 16 + m];
        v8f c = {bi, bi, bi, bi, bi, bi, bi, bi};
        #pragma unroll
        for (int kc = 0; kc < 4; ++kc) c = wmma_bf16(z1A[kc], load_bfragT(Wc2t, nt, kc, lane), c);
        #pragma unroll
        for (int r = 0; r < 8; ++r) my2[(r + roff) * 64 + nt * 16 + m] = fmaxf(c[r], 0.f);
    }
    wave_lds_fence();

    // ---- logits (64x2) + log_softmax, lanes 0..15 ----
    if (lane < 16) {
        float l0 = bc3[0], l1 = bc3[1];
        #pragma unroll 8
        for (int k = 0; k < 64; ++k) {
            float z = my2[m * 64 + k];
            l0 += z * Wc3[2 * k];
            l1 += z * Wc3[2 * k + 1];
        }
        float mx = fmaxf(l0, l1);
        float lse = mx + logf(__expf(l0 - mx) + __expf(l1 - mx));
        outm[(size_t)e * 2 + 0] = l0 - lse;
        outm[(size_t)e * 2 + 1] = l1 - lse;
    }
}

// ---------------- launch ----------------

static inline int cdiv(long long a, long long b) { return (int)((a + b - 1) / b); }

extern "C" void kernel_launch(void* const* d_in, const int* in_sizes, int n_in,
                              void* d_out, int out_size, void* d_ws, size_t ws_size,
                              hipStream_t stream) {
    const float* x   = (const float*)d_in[0];
    const int*   ei  = (const int*)d_in[1];
    const float* ea  = (const float*)d_in[2];
    const float* W1  = (const float*)d_in[3];
    const float* b1  = (const float*)d_in[4];
    const float* W2  = (const float*)d_in[5];
    const float* b2  = (const float*)d_in[6];
    const float* W3  = (const float*)d_in[7];
    const float* b3  = (const float*)d_in[8];
    const float* We1 = (const float*)d_in[9];
    const float* be1 = (const float*)d_in[10];
    const float* We2 = (const float*)d_in[11];
    const float* be2 = (const float*)d_in[12];
    const float* Wc1 = (const float*)d_in[13];
    const float* bc1 = (const float*)d_in[14];
    const float* Wc2 = (const float*)d_in[15];
    const float* bc2 = (const float*)d_in[16];
    const float* Wc3 = (const float*)d_in[17];
    const float* bc3 = (const float*)d_in[18];
    float* out = (float*)d_out;

    const int N = in_sizes[0] / 8;
    const int E = in_sizes[1] / 2;
    const int* row = ei;
    const int* col = ei + E;

    // workspace carve (256B aligned)
    char* p = (char*)d_ws;
    auto alloc = [&](size_t bytes) { char* r = p; p += (bytes + 255) & ~(size_t)255; return r; };
    float*  deg   = (float*)alloc((size_t)N * 4);
    float*  dinv  = (float*)alloc((size_t)N * 4);
    float*  bufA  = (float*)alloc((size_t)N * H * 4);   // out / h (fp32)
    float*  bufB  = (float*)alloc((size_t)N * H * 4);   // tmp (fp32)
    __bf16* h_bf  = (__bf16*)alloc((size_t)N * H * 2);
    __bf16* W2t   = (__bf16*)alloc((size_t)H * H * 2);
    __bf16* W3t   = (__bf16*)alloc((size_t)H * H * 2);
    __bf16* We2t  = (__bf16*)alloc((size_t)H * H * 2);
    __bf16* Wc1at = (__bf16*)alloc((size_t)H * H * 2);
    __bf16* Wc1bt = (__bf16*)alloc((size_t)H * H * 2);
    __bf16* Wc1ct = (__bf16*)alloc((size_t)H * H * 2);
    __bf16* Wc2t  = (__bf16*)alloc((size_t)H * 64 * 2);
    (void)ws_size; (void)n_in; (void)out_size;

    const int T = 256;
    // weight conversion / transpose
    k_cvt_transpose<<<cdiv(H * H, T), T, 0, stream>>>(W2, W2t, H, H);
    k_cvt_transpose<<<cdiv(H * H, T), T, 0, stream>>>(W3, W3t, H, H);
    k_cvt_transpose<<<cdiv(H * H, T), T, 0, stream>>>(We2, We2t, H, H);
    k_cvt_transpose<<<cdiv(H * H, T), T, 0, stream>>>(Wc1, Wc1at, H, H);
    k_cvt_transpose<<<cdiv(H * H, T), T, 0, stream>>>(Wc1 + H * H, Wc1bt, H, H);
    k_cvt_transpose<<<cdiv(H * H, T), T, 0, stream>>>(Wc1 + 2 * H * H, Wc1ct, H, H);
    k_cvt_transpose<<<cdiv(H * 64, T), T, 0, stream>>>(Wc2, Wc2t, H, 64);

    // degrees + symmetric norm
    k_deg_init<<<cdiv(N, T), T, 0, stream>>>(deg, N);
    k_deg_edges<<<cdiv(E, T), T, 0, stream>>>(col, deg, E);
    k_dinv<<<cdiv(N, T), T, 0, stream>>>(deg, dinv, N);

    const long long NH = (long long)N * H;
    // --- GCN layer 1 (K=8 input) ---
    k_gemm_in8<<<cdiv(NH, T), T, 0, stream>>>(x, W1, bufB, N);
    k_selfinit<<<cdiv(NH, T), T, 0, stream>>>(bufB, dinv, b1, bufA, N);
    k_scatter<<<cdiv((long long)E * 32, T), T, 0, stream>>>(bufB, dinv, row, col, bufA, E);
    k_relu_cvt<<<cdiv(NH, T), T, 0, stream>>>(bufA, h_bf, (int)NH);
    // --- GCN layer 2 ---
    k_node_gemm<<<N / 16, 256, 0, stream>>>(h_bf, W2t, bufB, N);
    k_selfinit<<<cdiv(NH, T), T, 0, stream>>>(bufB, dinv, b2, bufA, N);
    k_scatter<<<cdiv((long long)E * 32, T), T, 0, stream>>>(bufB, dinv, row, col, bufA, E);
    k_relu_cvt<<<cdiv(NH, T), T, 0, stream>>>(bufA, h_bf, (int)NH);
    // --- GCN layer 3 ---
    k_node_gemm<<<N / 16, 256, 0, stream>>>(h_bf, W3t, bufB, N);
    k_selfinit<<<cdiv(NH, T), T, 0, stream>>>(bufB, dinv, b3, bufA, N);
    k_scatter<<<cdiv((long long)E * 32, T), T, 0, stream>>>(bufB, dinv, row, col, bufA, E);
    k_relu_cvt<<<cdiv(NH, T), T, 0, stream>>>(bufA, h_bf, (int)NH);

    // --- fused edge encoder + classifier (WMMA-heavy) ---
    int tiles = E / 16;
    k_edge_head<<<cdiv(tiles, WAVES), 256, 0, stream>>>(
        row, col, ea, h_bf, We1, be1, We2t, be2,
        Wc1at, Wc1bt, Wc1ct, bc1, Wc2t, bc2, Wc3, bc3, out, E);
}
